// SelfAttentionBlock_9740985827671
// MI455X (gfx1250) — matrix operand order repaired and checked by reading, not compiled, not run on hardware
//
#include <hip/hip_runtime.h>

typedef __bf16 bf16;
typedef __attribute__((ext_vector_type(16))) __bf16 v16bf;
typedef __attribute__((ext_vector_type(8)))  __bf16 v8bf;
typedef __attribute__((ext_vector_type(8)))  float  v8f;
typedef __attribute__((ext_vector_type(4)))  unsigned int u32x4;
typedef __attribute__((ext_vector_type(8)))  int i32x8;
typedef __attribute__((ext_vector_type(4)))  int i32x4;

#define NEGV  (-4294967295.0f)
#define EPSV  (1e-8f)

__device__ __forceinline__ bf16 f2bf(float f) {
  unsigned u = __builtin_bit_cast(unsigned, f);
  u += 0x7FFFu + ((u >> 16) & 1u);               // round-to-nearest-even
  unsigned short s = (unsigned short)(u >> 16);
  return __builtin_bit_cast(bf16, s);
}

union AFrag  { v16bf v; v8bf h[2]; };
union Pack16 { v16bf v; v8bf h[2]; bf16 e[16]; };
union OFragB { v8bf v; bf16 e[8]; };

__device__ __forceinline__ v8f wmma_bf16f32(v16bf a, v16bf b, v8f c) {
  // (neg_a, A, neg_b, B, c_mod, C, reuse_a, reuse_b)
  return __builtin_amdgcn_wmma_f32_16x16x32_bf16(false, a, false, b, (short)0, c,
                                                 false, false);
}

// ---------------------------------------------------------------------------
// TDM: issue a 2D tile load (rows x 32 bf16) global -> LDS.
// LDS pad: after every 16 DWORDs (one 64B row) insert 4 DWORDs (16B)
//   -> LDS row pitch = 80B = 40 bf16 = LDST (matches WMMA fragment reads).
// D# layout per CDNA5 ISA.
// amdgpu-toolchain builtin: 6 args (u32x4, i32x8, i32x4, i32x4, i32x8, cpol).
// ---------------------------------------------------------------------------
__device__ __forceinline__ void tdm_load_tile(unsigned lds_off, const void* gptr,
                                              int rows, int rowlen,
                                              long long row_stride) {
  unsigned long long ga = (unsigned long long)gptr;
  u32x4 g0;
  g0[0] = 1u;                                            // count=1, user mode
  g0[1] = lds_off;                                       // lds_addr (bytes)
  g0[2] = (unsigned)(ga & 0xFFFFFFFFull);                // global_addr[31:0]
  g0[3] = (unsigned)((ga >> 32) & 0x1FFFFFFull)          // global_addr[56:32]
        | (2u << 30);                                    // type=2 ("image")
  i32x8 g1;
  g1[0] = 0x10000                                        // data_size=1 (2B)
        | (1 << 20)                                      // pad_enable
        | (3 << 22)                                      // pad_interval: 16 DW
        | (3 << 25);                                     // pad_amount: 4 DW
  g1[1] = (rowlen & 0xFFFF) << 16;                       // tensor_dim0 lo16
  g1[2] = ((rowlen >> 16) & 0xFFFF) | ((rows & 0xFFFF) << 16); // dim0 hi / dim1 lo
  g1[3] = ((rows >> 16) & 0xFFFF) | (32 << 16);          // dim1 hi / tile_dim0=32
  g1[4] = rows & 0xFFFF;                                 // tile_dim1=rows, tile_dim2=0
  g1[5] = (int)(row_stride & 0xFFFFFFFFll);              // dim0_stride lo32
  g1[6] = (int)((row_stride >> 32) & 0xFFFFll);          // dim0_stride hi16
  g1[7] = 0;
  i32x4 z4 = {0, 0, 0, 0};
  i32x8 z8 = {0, 0, 0, 0, 0, 0, 0, 0};
  __builtin_amdgcn_tensor_load_to_lds(g0, g1, z4, z4, z8, 0);
}

// ---------------------------------------------------------------------------
// Generic WMMA GEMM:  out[m,n] = act( scale * sum_k A[m,k]*W[n,k] + bias[n] )
//                                (+ res[m,n])
// Block tile 128(M) x 128(N), 8 waves, each wave owns 16 M-rows and all 128 N.
// Tiles double-buffered in LDS; B tile (and A tile when bf16) streamed by the
// Tensor Data Mover, f32 A converted via register staging. One barrier/k-step.
// ---------------------------------------------------------------------------
template <typename AT, typename OT, bool TRANS_OUT, bool RELU, bool HAS_BIAS,
          bool HAS_RES, bool CAUSAL_N, bool CAUSAL_K>
__global__ __launch_bounds__(256) void gemm_wmma(
    const AT* __restrict__ A, long long sAb, int lda,
    const bf16* __restrict__ W, long long sWb, int ldw,
    const float* __restrict__ bias,
    const float* __restrict__ res, long long sRb, int ldr,
    OT* __restrict__ out, long long sOb, int ldo,
    int M, int N, int K, float scale)
{
  constexpr int LDST = 40;                 // padded LDS row stride (bf16 elems)
  const int m0 = blockIdx.x * 128;
  const int n0 = blockIdx.y * 128;
  if (CAUSAL_N && n0 > m0 + 127) return;   // fully above causal diagonal

  const int b = blockIdx.z;
  A += (long long)b * sAb;
  W += (long long)b * sWb;
  out += (long long)b * sOb;
  if (HAS_RES) res += (long long)b * sRb;

  __shared__ bf16 As[2][128 * LDST];       // 2 x 10 KB
  __shared__ bf16 Bs[2][128 * LDST];       // 2 x 10 KB

  const int tid  = threadIdx.x;
  const int wave = tid >> 5;
  const int lane = tid & 31;
  const int half = lane >> 4;              // which 16-lane half
  const int l16  = lane & 15;
  const int mw   = wave * 16;              // wave's M-row group in tile

  const unsigned as_off[2] = { (unsigned)(unsigned long long)(uintptr_t)&As[0][0],
                               (unsigned)(unsigned long long)(uintptr_t)&As[1][0] };
  const unsigned bs_off[2] = { (unsigned)(unsigned long long)(uintptr_t)&Bs[0][0],
                               (unsigned)(unsigned long long)(uintptr_t)&Bs[1][0] };

  v8f acc[8] = {};                         // 16 x 128 f32 per wave

  const int srow = tid >> 1;               // f32-A staging: 2 threads / row
  const int skof = (tid & 1) * 16;

  int kend = K;
  if (CAUSAL_K) { int ce = m0 + 128; kend = ce < K ? ce : K; }

  float aRegF[16];

  auto load_nextA = [&](int kk) {          // only for AT == float
    const AT* ag = A + (long long)(m0 + srow) * lda + kk + skof;
#pragma unroll
    for (int i = 0; i < 16; ++i) aRegF[i] = (float)ag[i];
  };
  auto store_bufA = [&](int buf) {
    bf16* as = &As[buf][srow * LDST + skof];
    Pack16 t;
#pragma unroll
    for (int i = 0; i < 16; ++i) t.e[i] = f2bf(aRegF[i]);
    *(v8bf*)as = t.h[0]; *(v8bf*)(as + 8) = t.h[1];
  };
  auto issue_tdm = [&](int buf, int kk) {  // wave 0 only
    tdm_load_tile(bs_off[buf], W + (long long)n0 * ldw + kk, 128, ldw, ldw);
    if constexpr (sizeof(AT) == 2)
      tdm_load_tile(as_off[buf], A + (long long)m0 * lda + kk, 128, lda, lda);
  };

  // ---- prologue: fill buffer 0 ----
  if (wave == 0) issue_tdm(0, 0);
  if constexpr (sizeof(AT) == 4) { load_nextA(0); store_bufA(0); }
  if (wave == 0) __builtin_amdgcn_s_wait_tensorcnt(0);
  __syncthreads();

  int cur = 0;
  for (int k0 = 0; k0 < kend; k0 += 32) {
    const bool more = (k0 + 32) < kend;
    if (more) {
      if (wave == 0) issue_tdm(cur ^ 1, k0 + 32);   // DMA overlaps wmma below
      if constexpr (sizeof(AT) == 4) load_nextA(k0 + 32);
    }
    if constexpr (sizeof(AT) == 4) {
      if (k0 + 64 < kend)                            // L2 prefetch 2 steps ahead
        __builtin_prefetch(A + (long long)(m0 + srow) * lda + k0 + 64 + skof, 0, 3);
    }

    // A fragment: row (mw+l16), K halves split across lane groups
    AFrag a;
    const bf16* ar = &As[cur][(mw + l16) * LDST];
    a.h[0] = *(const v8bf*)(ar + half * 8);
    a.h[1] = *(const v8bf*)(ar + 16 + half * 8);

#pragma unroll
    for (int j = 0; j < 8; ++j) {
      const bf16* br = &Bs[cur][(j * 16 + l16) * LDST];
      AFrag bf;
      bf.h[0] = *(const v8bf*)(br + half * 8);
      bf.h[1] = *(const v8bf*)(br + 16 + half * 8);
      v16bf bv = bf.v;
      acc[j] = wmma_bf16f32(a.v, bv, acc[j]);
    }

    if (more) {
      if constexpr (sizeof(AT) == 4) store_bufA(cur ^ 1);
      if (wave == 0) __builtin_amdgcn_s_wait_tensorcnt(0);
      __syncthreads();                               // one barrier per k-step
      cur ^= 1;
    }
  }

  // ---- epilogue ----
#pragma unroll
  for (int j = 0; j < 8; ++j) {
    const int n  = n0 + j * 16 + l16;
    const int mb = m0 + mw + half * 8;
    float bj = 0.f;
    if (HAS_BIAS) bj = bias[n];
    if constexpr (TRANS_OUT) {
      OFragB o;
#pragma unroll
      for (int r = 0; r < 8; ++r) {
        float v = acc[j][r] * scale + bj;
        if (RELU) v = v > 0.f ? v : 0.f;
        o.e[r] = f2bf(v);
      }
      *(v8bf*)(out + (long long)n * ldo + mb) = o.v;   // contiguous along m
    } else {
#pragma unroll
      for (int r = 0; r < 8; ++r) {
        float v = acc[j][r] * scale + bj;
        if (RELU) v = v > 0.f ? v : 0.f;
        if (HAS_RES) v += res[(long long)(mb + r) * ldr + n];
        if constexpr (sizeof(OT) == 4) out[(long long)(mb + r) * ldo + n] = (OT)v;
        else                           out[(long long)(mb + r) * ldo + n] = f2bf(v);
      }
    }
  }
}

// ---------------------------------------------------------------------------
// LayerNorm over C=1024 (unbiased var, ddof=1), optional mask outputs.
// One 256-thread block per token row.
// ---------------------------------------------------------------------------
template <bool MASKS>
__global__ __launch_bounds__(256) void ln_kernel(
    const float* __restrict__ x, const float* __restrict__ g,
    const float* __restrict__ beta, float* __restrict__ out,
    float* __restrict__ keymask, float* __restrict__ qmask)
{
  const int C = 1024;
  const long long row = blockIdx.x;
  const int t = threadIdx.x;
  const float* xr = x + row * C;
  float v[4];
  float s = 0.f, sq = 0.f;
#pragma unroll
  for (int i = 0; i < 4; ++i) {
    v[i] = xr[t + i * 256];
    s += v[i]; sq += v[i] * v[i];
  }
  __shared__ float r1[256], r2[256];
  r1[t] = s; r2[t] = sq; __syncthreads();
  for (int st = 128; st > 0; st >>= 1) {
    if (t < st) { r1[t] += r1[t + st]; r2[t] += r2[t + st]; }
    __syncthreads();
  }
  const float tot = r1[0], totsq = r2[0];
  __syncthreads();
  const float mean = tot * (1.f / 1024.f);
  const float var  = (totsq - tot * mean) * (1.f / 1023.f);  // ddof=1
  const float inv  = rsqrtf(var + EPSV);
  float qs = 0.f;
  float* orow = out + row * C;
#pragma unroll
  for (int i = 0; i < 4; ++i) {
    const int c = t + i * 256;
    const float q = g[c] * (v[i] - mean) * inv + beta[c];
    orow[c] = q; qs += q;
  }
  if (MASKS) {
    r1[t] = qs; __syncthreads();
    for (int st = 128; st > 0; st >>= 1) {
      if (t < st) r1[t] += r1[t + st];
      __syncthreads();
    }
    if (t == 0) {
      keymask[row] = (tot   != 0.f) ? 1.f : 0.f;  // sign(|sum(x)|)
      qmask[row]   = (r1[0] != 0.f) ? 1.f : 0.f;  // sign(|sum(queries)|)
    }
  }
}

// ---------------------------------------------------------------------------
// Row softmax with key/causal masks; writes P = softmax * qmask as bf16.
// One 256-thread block per (b, q) row of length T=2048.
// ---------------------------------------------------------------------------
__global__ __launch_bounds__(256) void softmax_kernel(
    const float* __restrict__ scores, const float* __restrict__ keymask,
    const float* __restrict__ qmask, bf16* __restrict__ P)
{
  const int T = 2048;
  const long long row = blockIdx.x;      // b*T + q
  const int q = (int)(row & (T - 1));
  const long long bb = row >> 11;
  const int t = threadIdx.x;
  const float* srow = scores + row * (long long)T;
  const float* km = keymask + bb * T;
  float sv[8];
  float mx = NEGV;
#pragma unroll
  for (int i = 0; i < 8; ++i) {
    const int k = t + i * 256;
    const float s = (k <= q && km[k] != 0.f) ? srow[k] : NEGV;
    sv[i] = s; mx = fmaxf(mx, s);
  }
  __shared__ float red[256];
  red[t] = mx; __syncthreads();
  for (int st = 128; st > 0; st >>= 1) {
    if (t < st) red[t] = fmaxf(red[t], red[t + st]);
    __syncthreads();
  }
  mx = red[0]; __syncthreads();
  float ev[8];
  float se = 0.f;
#pragma unroll
  for (int i = 0; i < 8; ++i) { ev[i] = __expf(sv[i] - mx); se += ev[i]; }
  red[t] = se; __syncthreads();
  for (int st = 128; st > 0; st >>= 1) {
    if (t < st) red[t] += red[t + st];
    __syncthreads();
  }
  const float inv = qmask[row] / red[0];
  bf16* prow = P + row * (long long)T;
#pragma unroll
  for (int i = 0; i < 8; ++i) prow[t + i * 256] = f2bf(ev[i] * inv);
}

// fp32 -> bf16 weight conversion (1M elements, 4/thread)
__global__ __launch_bounds__(256) void cvt_kernel(const float* __restrict__ in,
                                                  bf16* __restrict__ out)
{
  const long long i = ((long long)blockIdx.x * 256 + threadIdx.x) * 4;
#pragma unroll
  for (int j = 0; j < 4; ++j) out[i + j] = f2bf(in[i + j]);
}

// ---------------------------------------------------------------------------
extern "C" void kernel_launch(void* const* d_in, const int* in_sizes, int n_in,
                              void* d_out, int out_size, void* d_ws, size_t ws_size,
                              hipStream_t stream)
{
  (void)in_sizes; (void)n_in; (void)out_size; (void)ws_size;
  const int B = 8, T = 2048, C = 1024;
  const long long NT = (long long)B * T;

  const float* x       = (const float*)d_in[0];
  const float* fc1_w   = (const float*)d_in[1];
  const float* fc1_b   = (const float*)d_in[2];
  const float* fc2_w   = (const float*)d_in[3];
  const float* fc2_b   = (const float*)d_in[4];
  const float* fc3_w   = (const float*)d_in[5];
  const float* fc3_b   = (const float*)d_in[6];
  const float* ln1_g   = (const float*)d_in[7];
  const float* ln1_b   = (const float*)d_in[8];
  const float* ln2_g   = (const float*)d_in[9];
  const float* ln2_b   = (const float*)d_in[10];
  const float* conv1_w = (const float*)d_in[11];
  const float* conv1_b = (const float*)d_in[12];
  const float* conv2_w = (const float*)d_in[13];
  const float* conv2_b = (const float*)d_in[14];
  float* out = (float*)d_out;

  char* p = (char*)d_ws;
  auto alloc = [&](size_t bytes) -> char* {
    char* r = p; p += (bytes + 255) & ~(size_t)255; return r;
  };
  const size_t szW = (size_t)C * C * sizeof(bf16);
  bf16* wq   = (bf16*)alloc(szW);
  bf16* wk   = (bf16*)alloc(szW);
  bf16* wv   = (bf16*)alloc(szW);
  bf16* wc1  = (bf16*)alloc(szW);
  bf16* wc2  = (bf16*)alloc(szW);
  float* queries = (float*)alloc((size_t)NT * C * 4);
  float* keymask = (float*)alloc((size_t)NT * 4);
  float* qmask   = (float*)alloc((size_t)NT * 4);
  bf16* Qb = (bf16*)alloc((size_t)NT * C * 2);
  bf16* Kb = (bf16*)alloc((size_t)NT * C * 2);
  bf16* Vt = (bf16*)alloc((size_t)NT * C * 2);              // [B][C][T]
  float* scoresf = (float*)alloc((size_t)B * T * T * 4);    // 128MB, reused below
  bf16*  P       = (bf16*)alloc((size_t)B * T * T * 2);
  float* attnout = (float*)alloc((size_t)NT * C * 4);
  // scores region is dead after softmax/pv -> reuse for z and h
  float* z  = scoresf;
  bf16*  hb = (bf16*)((char*)scoresf + (size_t)NT * C * 4);

  // 1) weights fp32 -> bf16
  cvt_kernel<<<dim3(1024), 256, 0, stream>>>(fc1_w,   wq);
  cvt_kernel<<<dim3(1024), 256, 0, stream>>>(fc2_w,   wk);
  cvt_kernel<<<dim3(1024), 256, 0, stream>>>(fc3_w,   wv);
  cvt_kernel<<<dim3(1024), 256, 0, stream>>>(conv1_w, wc1);
  cvt_kernel<<<dim3(1024), 256, 0, stream>>>(conv2_w, wc2);

  // 2) LN1 + masks
  ln_kernel<true><<<dim3((unsigned)NT), 256, 0, stream>>>(
      x, ln1_g, ln1_b, queries, keymask, qmask);

  // 3) projections: Q = LN(x)@fc1^T, K = x@fc2^T, V^T = (x@fc3^T)^T
  gemm_wmma<float, bf16, false, false, true, false, false, false>
      <<<dim3(128, 8, 1), 256, 0, stream>>>(
          queries, 0, C, wq, 0, C, fc1_b, nullptr, 0, 0,
          Qb, 0, C, 16384, 1024, 1024, 1.f);
  gemm_wmma<float, bf16, false, false, true, false, false, false>
      <<<dim3(128, 8, 1), 256, 0, stream>>>(
          x, 0, C, wk, 0, C, fc2_b, nullptr, 0, 0,
          Kb, 0, C, 16384, 1024, 1024, 1.f);
  gemm_wmma<float, bf16, true, false, true, false, false, false>
      <<<dim3(16, 8, 8), 256, 0, stream>>>(
          x, (long long)T * C, C, wv, 0, C, fc3_b, nullptr, 0, 0,
          Vt, (long long)C * T, T, 2048, 1024, 1024, 1.f);

  // 4) scores = Q K^T / 32 (skip blocks above causal diagonal)
  gemm_wmma<bf16, float, false, false, false, false, true, false>
      <<<dim3(16, 16, 8), 256, 0, stream>>>(
          Qb, (long long)T * C, C, Kb, (long long)T * C, C, nullptr,
          nullptr, 0, 0, scoresf, (long long)T * T, T,
          2048, 2048, 1024, 0.03125f);

  // 5) masked softmax -> P (bf16, qmask folded in)
  softmax_kernel<<<dim3((unsigned)NT), 256, 0, stream>>>(scoresf, keymask, qmask, P);

  // 6) attnout = P @ V + queries (causal K-limit)
  gemm_wmma<bf16, float, false, false, false, true, false, true>
      <<<dim3(16, 8, 8), 256, 0, stream>>>(
          P, (long long)T * T, T, Vt, (long long)C * T, T, nullptr,
          queries, (long long)T * C, C, attnout, (long long)T * C, C,
          2048, 1024, 2048, 1.f);

  // 7) z = LN2(attnout)
  ln_kernel<false><<<dim3((unsigned)NT), 256, 0, stream>>>(
      attnout, ln2_g, ln2_b, z, nullptr, nullptr);

  // 8) h = relu(z @ conv1^T + b1)  (bf16)
  gemm_wmma<float, bf16, false, true, true, false, false, false>
      <<<dim3(128, 8, 1), 256, 0, stream>>>(
          z, 0, C, wc1, 0, C, conv1_b, nullptr, 0, 0,
          hb, 0, C, 16384, 1024, 1024, 1.f);

  // 9) y = h @ conv2^T + b2 + z  (fp32 -> d_out)
  gemm_wmma<bf16, float, false, false, true, true, false, false>
      <<<dim3(128, 8, 1), 256, 0, stream>>>(
          hb, 0, C, wc2, 0, C, conv2_b, z, 0, C,
          out, 0, C, 16384, 1024, 1024, 1.f);
}
